// VanillaCGN_57251914056250
// MI455X (gfx1250) — matrix operand order repaired and checked by reading, not compiled, not run on hardware
//
#include <hip/hip_runtime.h>

typedef _Float16 h16;
typedef __attribute__((ext_vector_type(16))) _Float16 v16h;
typedef __attribute__((ext_vector_type(8)))  _Float16 v8h;
typedef __attribute__((ext_vector_type(8)))  float    v8f;
typedef __attribute__((ext_vector_type(4)))  float    v4f;
typedef __attribute__((ext_vector_type(4)))  unsigned int v4u;

#define GN 16384
#define GD 64

// ---- CDNA5 async global->LDS copy (ASYNCcnt tracked, no VGPR data path) ----
__device__ __forceinline__ void async_ld_b128(const h16* gptr, const h16* lptr) {
    unsigned loff = (unsigned)(size_t)lptr;   // flat LDS addr low 32 bits = LDS offset
    asm volatile("global_load_async_to_lds_b128 %0, %1, off"
                 :: "v"(loff), "v"(gptr)
                 : "memory");
}
__device__ __forceinline__ void wait_async0() {
    asm volatile("s_wait_asynccnt 0x0" ::: "memory");
}

// ---------------------------------------------------------------- zero deg
__global__ __launch_bounds__(256) void zero_deg_k(float* __restrict__ deg) {
    int i = blockIdx.x * 256 + threadIdx.x;
    if (i < GN) deg[i] = 0.f;
}

// ---------------------------------------------------------------- prep:
// stream adj f32 (1 GB) once; emit transposed 1-bit mask (32 MB, L2-resident)
// and fused column sums deg[c] = sum_r adj[r][c].
__global__ __launch_bounds__(256) void prep_adj_k(const float* __restrict__ adj,
                                                  unsigned char* __restrict__ abits,
                                                  float* __restrict__ deg) {
    const int c  = blockIdx.x * 256 + threadIdx.x;   // column (= adjT row)
    const int r0 = blockIdx.y * 64;                  // row chunk
    unsigned long long mask = 0ull;
    float s = 0.f;
    const float* p = adj + (size_t)r0 * GN + c;      // coalesced across threads
#pragma unroll 8
    for (int rr = 0; rr < 64; ++rr) {
        float v = p[(size_t)rr * GN];
        s += v;
        mask |= (unsigned long long)(v != 0.f) << rr;  // bit k -> adj[k][c]
    }
    *(unsigned long long*)(abits + (size_t)c * 2048 + (r0 >> 3)) = mask;
    atomicAdd(deg + c, s);
}

// ---------------------------------------------------------------- input proj:
// hT[j][i] = (x[i] @ U0)[j] + b0[j], fp16 transposed layout [64][16384]
__global__ __launch_bounds__(256) void proj_k(const float* __restrict__ x,
                                              const float* __restrict__ U0,
                                              const float* __restrict__ b0,
                                              h16* __restrict__ hT) {
    const int i = blockIdx.x * 256 + threadIdx.x;
    float xr[GD];
#pragma unroll
    for (int k = 0; k < GD; ++k) xr[k] = x[(size_t)i * GD + k];
    for (int j = 0; j < GD; ++j) {
        float a = b0[j];
#pragma unroll
        for (int k = 0; k < GD; ++k) a = fmaf(xr[k], U0[k * GD + j], a);
        hT[(size_t)j * GN + i] = (h16)a;   // coalesced across i per j
    }
}

// ---------------------------------------------------------------- GCN layer:
// hT_out = relu( ((adjT @ h) / deg) @ U^T ) , matmuls via v_wmma_f32_16x16x32_f16
__global__ __launch_bounds__(128) void gcn_layer_k(const unsigned char* __restrict__ abits,
                                                   const h16* __restrict__ hT_in,
                                                   const float* __restrict__ deg,
                                                   const float* __restrict__ U,
                                                   h16* __restrict__ hT_out) {
    __shared__ h16 lut[256 * 8];        // byte -> 8 fp16 (0/1) expansion LUT
    __shared__ h16 hbuf[2][64 * 520];   // double-buffered 64x512 h chunk (1040 B row stride)
    __shared__ h16 aggs[4][16 * 72];    // per-wave 16x64 agg tile (144 B row stride)

    const int tid  = threadIdx.x;
    const int wave = tid >> 5;
    const int lane = tid & 31;
    const int nl   = lane & 15;
    const int hi   = lane >> 4;

    // build LUT (bit j of byte v -> half j)
    for (int v = tid; v < 256; v += 128) {
#pragma unroll
        for (int j = 0; j < 8; ++j) lut[v * 8 + j] = (h16)((v >> j) & 1);
    }

    const int m0 = blockIdx.x * 64 + wave * 16;               // wave's 16 output rows
    const unsigned char* arow = abits + (size_t)(m0 + nl) * 2048;

    v8f acc[4] = {};                                          // 16x64 f32 accumulators

    // prime buffer 0 with async copies (32 x b128 per thread, ASYNCcnt <= 32)
    for (int i = tid; i < 64 * 64; i += 128) {
        int n = i >> 6, j = i & 63;
        async_ld_b128(hT_in + (size_t)n * GN + 0 * 512 + j * 8,
                      &hbuf[0][n * 520 + j * 8]);
    }
    wait_async0();
    __syncthreads();

    for (int kc = 0; kc < 32; ++kc) {                         // K chunks of 512
        const int cur = kc & 1;
        if (kc < 31) {                                        // prefetch next chunk (async)
            for (int i = tid; i < 64 * 64; i += 128) {
                int n = i >> 6, j = i & 63;
                async_ld_b128(hT_in + (size_t)n * GN + (kc + 1) * 512 + j * 8,
                              &hbuf[cur ^ 1][n * 520 + j * 8]);
            }
        }

        // hoist this chunk's A bits: 64 B per lane -> 16 VGPRs, off the WMMA chain
        v4u ab[4];
        {
            const v4u* ap = (const v4u*)(arow + kc * 64);
#pragma unroll
            for (int q = 0; q < 4; ++q) ab[q] = ap[q];
        }

#pragma unroll
        for (int ks = 0; ks < 16; ++ks) {                     // K steps of 32
            unsigned int bits = ab[ks >> 2][ks & 3];
            unsigned int blo = (bits >> (hi * 8)) & 0xFFu;          // K k0+8*hi .. +7
            unsigned int bhi = (bits >> (16 + hi * 8)) & 0xFFu;     // K k0+16+8*hi .. +7
            v8h alo = *(const v8h*)&lut[blo * 8];
            v8h ahi = *(const v8h*)&lut[bhi * 8];
            v16h A = __builtin_shufflevector(alo, ahi, 0,1,2,3,4,5,6,7,
                                                       8,9,10,11,12,13,14,15);
            const int kb = ks * 32 + hi * 8;
#pragma unroll
            for (int nt = 0; nt < 4; ++nt) {
                const h16* bp = &hbuf[cur][(nt * 16 + nl) * 520 + kb];
                v8h b0v = *(const v8h*)bp;
                v8h b1v = *(const v8h*)(bp + 16);
                v16h B = __builtin_shufflevector(b0v, b1v, 0,1,2,3,4,5,6,7,
                                                           8,9,10,11,12,13,14,15);
                acc[nt] = __builtin_amdgcn_wmma_f32_16x16x32_f16(
                    false, A, false, B, (short)0, acc[nt], false, false);
            }
        }

        if (kc < 31) wait_async0();   // own async writes done (overlapped with WMMAs)
        __syncthreads();              // all waves: done reading cur, next buffer ready
    }

    // ---- scale by 1/deg, drop agg tile (fp16) into LDS for re-layout D->A
    v4f dga = *(const v4f*)(deg + m0 + hi * 8);
    v4f dgb = *(const v4f*)(deg + m0 + hi * 8 + 4);
#pragma unroll
    for (int nt = 0; nt < 4; ++nt) {
#pragma unroll
        for (int r = 0; r < 8; ++r) {
            float dv  = (r < 4) ? dga[r] : dgb[r - 4];
            float val = acc[nt][r] / dv;                 // D-layout: m = hi*8+r, n = nt*16+nl
            aggs[wave][(hi * 8 + r) * 72 + nt * 16 + nl] = (h16)val;
        }
    }
    __syncthreads();

    // ---- second matmul: hout[m][j] = relu( sum_k agg[m][k] * U[j][k] )
    v16h Ag[2];
#pragma unroll
    for (int kk = 0; kk < 2; ++kk) {
        const h16* ap = &aggs[wave][nl * 72 + kk * 32 + hi * 16];
        v8h a0 = *(const v8h*)ap;          // K kk*32 + hi*16 .. +7
        v8h a1 = *(const v8h*)(ap + 8);    // K kk*32 + hi*16 + 8 .. +15
        Ag[kk] = __builtin_shufflevector(a0, a1, 0,1,2,3,4,5,6,7,
                                                 8,9,10,11,12,13,14,15);
    }
#pragma unroll
    for (int jt = 0; jt < 4; ++jt) {
        v8f o = {};
#pragma unroll
        for (int kk = 0; kk < 2; ++kk) {
            // B[k][j] = U[j][k]: lane column j = jt*16+nl, K contiguous in U row
            const float* up = U + (size_t)(jt * 16 + nl) * 64 + kk * 32 + hi * 16;
            v4f u0 = *(const v4f*)up;
            v4f u1 = *(const v4f*)(up + 4);
            v4f u2 = *(const v4f*)(up + 8);
            v4f u3 = *(const v4f*)(up + 12);
            v8h blo, bhi2;
#pragma unroll
            for (int t = 0; t < 4; ++t) {
                blo[t]      = (h16)u0[t];
                blo[t + 4]  = (h16)u1[t];
                bhi2[t]     = (h16)u2[t];
                bhi2[t + 4] = (h16)u3[t];
            }
            v16h Bu = __builtin_shufflevector(blo, bhi2, 0,1,2,3,4,5,6,7,
                                                         8,9,10,11,12,13,14,15);
            o = __builtin_amdgcn_wmma_f32_16x16x32_f16(
                false, Ag[kk], false, Bu, (short)0, o, false, false);
        }
        v8h outp;
#pragma unroll
        for (int r = 0; r < 8; ++r) outp[r] = (h16)fmaxf(o[r], 0.f);
        // D-layout: lane holds fixed j, 8 consecutive m -> contiguous b128 store in hT
        *(v8h*)(hT_out + (size_t)(jt * 16 + nl) * GN + m0 + hi * 8) = outp;
    }
}

// ---------------------------------------------------------------- readout:
// g = mean_i h[i], out = P @ relu(Q @ g)
__global__ __launch_bounds__(256) void readout_k(const h16* __restrict__ hT,
                                                 const float* __restrict__ Q,
                                                 const float* __restrict__ P,
                                                 float* __restrict__ out) {
    __shared__ float part[256];
    __shared__ float gsh[64];
    const int tid = threadIdx.x;
    const int j   = tid & 63;
    const int seg = tid >> 6;
    float s = 0.f;
    const v8h* row = (const v8h*)(hT + (size_t)j * GN + seg * 4096);
    for (int i = 0; i < 512; ++i) {
        v8h v = row[i];
#pragma unroll
        for (int e = 0; e < 8; ++e) s += (float)v[e];
    }
    part[tid] = s;
    __syncthreads();
    if (tid < 64)
        gsh[tid] = (part[tid] + part[tid + 64] + part[tid + 128] + part[tid + 192])
                   * (1.f / 16384.f);
    __syncthreads();
    if (tid < 64) {
        float a = 0.f;
        for (int jj = 0; jj < 64; ++jj) a = fmaf(Q[tid * 64 + jj], gsh[jj], a);
        part[tid] = P[tid] * fmaxf(a, 0.f);
    }
    __syncthreads();
    if (tid == 0) {
        float a = 0.f;
        for (int q = 0; q < 64; ++q) a += part[q];
        out[0] = a;
    }
}

// ---------------------------------------------------------------- launch
extern "C" void kernel_launch(void* const* d_in, const int* in_sizes, int n_in,
                              void* d_out, int out_size, void* d_ws, size_t ws_size,
                              hipStream_t stream) {
    const float* x   = (const float*)d_in[0];
    const float* adj = (const float*)d_in[1];
    const float* U0  = (const float*)d_in[2];
    const float* b0  = (const float*)d_in[3];
    const float* Us  = (const float*)d_in[4];
    const float* Q   = (const float*)d_in[5];
    const float* P   = (const float*)d_in[6];

    char* ws = (char*)d_ws;
    float*         deg   = (float*)ws;                                  // 64 KB
    unsigned char* abits = (unsigned char*)(ws + 65536);                // 32 MB (L2-resident)
    h16*           hA    = (h16*)(ws + 65536 + (size_t)GN * 2048);      // 2 MB
    h16*           hB    = hA + (size_t)GD * GN;                        // 2 MB

    zero_deg_k<<<64, 256, 0, stream>>>(deg);
    prep_adj_k<<<dim3(64, 256), 256, 0, stream>>>(adj, abits, deg);
    proj_k<<<64, 256, 0, stream>>>(x, U0, b0, hA);

    gcn_layer_k<<<256, 128, 0, stream>>>(abits, hA, deg, Us,            hB);
    gcn_layer_k<<<256, 128, 0, stream>>>(abits, hB, deg, Us + 4096,     hA);
    gcn_layer_k<<<256, 128, 0, stream>>>(abits, hA, deg, Us + 2 * 4096, hB);

    readout_k<<<1, 256, 0, stream>>>(hB, Q, P, (float*)d_out);
}
